// Block_13898514170126
// MI455X (gfx1250) — matrix operand order repaired and checked by reading, not compiled, not run on hardware
//
#include <hip/hip_runtime.h>

// ---------------------------------------------------------------------------
// Transformer block (B=2, T=2048, C=1024, H=16, hd=64, MLP 4x) for gfx1250.
// All GEMMs + attention run on v_wmma_f32_16x16x32_bf16 (fp32 accumulate).
// GEMM tiles staged via GLOBAL_LOAD_ASYNC_TO_LDS (ASYNCcnt) with double
// buffering when the toolchain exposes the builtin.
// ---------------------------------------------------------------------------

#define T_LEN   2048
#define EMBED   1024
#define NHEAD   16
#define HDIM    64
#define BT      4096          // B * T rows
#define C3      3072          // 3 * EMBED
#define HMLP    4096          // 4 * EMBED

#if defined(__has_builtin)
#if __has_builtin(__builtin_amdgcn_global_load_async_to_lds_b128)
#define HAVE_ASYNC_LDS 1
#endif
#endif

#define AS_GLOBAL __attribute__((address_space(1)))
#define AS_LOCAL  __attribute__((address_space(3)))

typedef __attribute__((ext_vector_type(16))) __bf16 v16bf;
typedef __attribute__((ext_vector_type(8)))  float  v8f;
typedef int v4i __attribute__((vector_size(16)));   // matches builtin proto V4i

union BF16x16 {
    v16bf         v;
    unsigned short u16[16];
    unsigned int   u32[8];
};

__device__ __forceinline__ v8f wmma_bf16(v16bf a, v16bf b, v8f c) {
    return __builtin_amdgcn_wmma_f32_16x16x32_bf16(
        /*neg_a=*/false, a, /*neg_b=*/false, b,
        /*c_mod=*/(short)0, c, /*reuse_a=*/false, /*reuse_b=*/false);
}

// A-matrix (16x32 bf16) per-lane K offset for vgpr-pair i (contiguous):
// lanes 0-15: V0..7 -> K 0..15 ; lanes 16-31: K 16..31
__device__ __forceinline__ int koffA(int i, int hl) {
    return hl * 16 + 2 * i;
}

__device__ __forceinline__ unsigned short f32_to_bf16(float f) {
    unsigned int u = __float_as_uint(f);
    unsigned int r = u + 0x7FFFu + ((u >> 16) & 1u);   // round-to-nearest-even
    return (unsigned short)(r >> 16);
}

#if defined(HAVE_ASYNC_LDS)
// 16B cache->LDS async copy, per-lane addresses, tracked by ASYNCcnt.
__device__ __forceinline__ void async_ld16(const unsigned short* g,
                                           unsigned short* l) {
    __builtin_amdgcn_global_load_async_to_lds_b128(
        (AS_GLOBAL v4i*)g, (AS_LOCAL v4i*)l, 0, 0);
}
__device__ __forceinline__ void wait_async0() {
#if __has_builtin(__builtin_amdgcn_s_wait_asynccnt)
    __builtin_amdgcn_s_wait_asynccnt(0);
#else
    asm volatile("s_wait_asynccnt 0" ::: "memory");
#endif
}
#endif

// ---------------------------------------------------------------------------
// fp32 -> bf16 elementwise cast (weights)
// ---------------------------------------------------------------------------
__global__ void cast_bf16_kernel(const float* __restrict__ in,
                                 unsigned short* __restrict__ out, int n) {
    int i = blockIdx.x * blockDim.x + threadIdx.x;
    int stride = gridDim.x * blockDim.x;
    for (; i < n; i += stride) out[i] = f32_to_bf16(in[i]);
}

// ---------------------------------------------------------------------------
// LayerNorm over last dim (1024) -> bf16 output. One block (256 thr) per row.
// ---------------------------------------------------------------------------
__global__ __launch_bounds__(256) void ln_bf16_kernel(
    const float* __restrict__ x, const float* __restrict__ scale,
    const float* __restrict__ shift, unsigned short* __restrict__ out) {
    const int row = blockIdx.x, tid = threadIdx.x;
    const float4 xr = *(const float4*)(x + (size_t)row * EMBED + tid * 4);
    float s  = xr.x + xr.y + xr.z + xr.w;
    float s2 = xr.x * xr.x + xr.y * xr.y + xr.z * xr.z + xr.w * xr.w;
    #pragma unroll
    for (int m = 16; m >= 1; m >>= 1) {
        s  += __shfl_xor(s,  m, 32);
        s2 += __shfl_xor(s2, m, 32);
    }
    __shared__ float ps[8], ps2[8];
    const int w = tid >> 5, lane = tid & 31;
    if (lane == 0) { ps[w] = s; ps2[w] = s2; }
    __syncthreads();
    float S = 0.f, S2 = 0.f;
    #pragma unroll
    for (int i = 0; i < 8; ++i) { S += ps[i]; S2 += ps2[i]; }
    const float mu  = S * (1.0f / EMBED);
    const float var = S2 * (1.0f / EMBED) - mu * mu;
    const float rs  = rsqrtf(var + 1e-5f);
    float xv[4] = {xr.x, xr.y, xr.z, xr.w};
    #pragma unroll
    for (int j = 0; j < 4; ++j) {
        const int c = tid * 4 + j;
        const float y = (xv[j] - mu) * rs * scale[c] + shift[c];
        out[(size_t)row * EMBED + c] = f32_to_bf16(y);
    }
}

// ---------------------------------------------------------------------------
// Residual add: c = a + b (fp32, float4 vectorized)
// ---------------------------------------------------------------------------
__global__ void resid_kernel(const float* __restrict__ a,
                             const float* __restrict__ b,
                             float* __restrict__ c) {
    const size_t i = ((size_t)blockIdx.x * blockDim.x + threadIdx.x) * 4;
    const float4 av = *(const float4*)(a + i);
    const float4 bv = *(const float4*)(b + i);
    float4 cv; cv.x = av.x + bv.x; cv.y = av.y + bv.y;
    cv.z = av.z + bv.z; cv.w = av.w + bv.w;
    *(float4*)(c + i) = cv;
}

// ---------------------------------------------------------------------------
// Per-K-tile fragment load + 2x WMMA (shared by both staging paths).
// ---------------------------------------------------------------------------
__device__ __forceinline__ void gemm_step(const unsigned short* __restrict__ Asb,
                                          const unsigned short* __restrict__ Bsb,
                                          int wm, int wn, int hl, int n16,
                                          v8f& acc0, v8f& acc1) {
    BF16x16 af, bf0, bf1;
    #pragma unroll
    for (int i = 0; i < 8; ++i)
        af.u32[i] = *(const unsigned int*)&Asb[(wm * 16 + n16) * 40 + koffA(i, hl)];
    #pragma unroll
    for (int i = 0; i < 8; ++i) {
        const int kk = hl * 16 + 2 * i;              // B 32x16 layout
        const int nc = wn * 32 + n16;
        bf0.u16[2 * i]     = Bsb[kk * 72 + nc];
        bf0.u16[2 * i + 1] = Bsb[(kk + 1) * 72 + nc];
        bf1.u16[2 * i]     = Bsb[kk * 72 + nc + 16];
        bf1.u16[2 * i + 1] = Bsb[(kk + 1) * 72 + nc + 16];
    }
    acc0 = wmma_bf16(af.v, bf0.v, acc0);
    acc1 = wmma_bf16(af.v, bf1.v, acc1);
}

// ---------------------------------------------------------------------------
// Tiled bf16 WMMA GEMM: D[M,N] = A[M,K] @ B[K,N] + bias (+resid) (ReLU opt).
// Block tile 64x64, BK=32, 256 threads = 8 waves (4 M-groups x 2 N-groups),
// each wave: one 16x32 A-frag, two B-frags, two fp32 16x16 accumulators.
// Double-buffered async LDS staging when available.
// ---------------------------------------------------------------------------
template <bool RELU, bool OUT_BF16, bool RESID>
__global__ __launch_bounds__(256, 2) void gemm_bf16_kernel(
    const unsigned short* __restrict__ A, const unsigned short* __restrict__ B,
    const float* __restrict__ bias, const float* __restrict__ resid,
    float* __restrict__ outF, unsigned short* __restrict__ outB,
    int M, int N, int K) {
    __shared__ unsigned short As[2][64 * 40];  // 64 rows x 32 K, stride 40
    __shared__ unsigned short Bs[2][32 * 72];  // 32 K   x 64 N, stride 72

    const int tid = threadIdx.x;
    const int lane = tid & 31, w = tid >> 5;
    const int hl = lane >> 4, n16 = lane & 15;
    const int wm = w & 3, wn = w >> 2;
    const int bm = blockIdx.y * 64, bn = blockIdx.x * 64;

    const int arow = tid >> 2, ak = (tid & 3) * 8;   // A: 4 thr/row, 8 elem ea
    const int brow = tid >> 3, bnc = (tid & 7) * 8;  // B: 8 thr/row, 8 elem ea
    const unsigned short* Ag = A + (size_t)(bm + arow) * K + ak;
    const unsigned short* Bg = B + (size_t)brow * N + bn + bnc;

    v8f acc0 = {}, acc1 = {};

#if defined(HAVE_ASYNC_LDS)
    // ---- double-buffered async pipeline (ASYNCcnt) ----
    async_ld16(Ag, &As[0][arow * 40 + ak]);
    async_ld16(Bg, &Bs[0][brow * 72 + bnc]);
    int cur = 0;
    for (int k0 = 0; k0 < K; k0 += 32) {
        wait_async0();          // own-wave tile landed
        __syncthreads();        // everyone's tile landed, prev reads retired
        if (k0 + 32 < K) {      // prefetch next tile into the other buffer
            async_ld16(Ag + k0 + 32, &As[cur ^ 1][arow * 40 + ak]);
            async_ld16(Bg + (size_t)(k0 + 32) * N, &Bs[cur ^ 1][brow * 72 + bnc]);
        }
        gemm_step(As[cur], Bs[cur], wm, wn, hl, n16, acc0, acc1);
        cur ^= 1;
    }
#else
    // ---- synchronous fallback staging ----
    for (int k0 = 0; k0 < K; k0 += 32) {
        const uint4 av = *(const uint4*)(Ag + k0);
        const uint4 bv = *(const uint4*)(Bg + (size_t)k0 * N);
        __syncthreads();
        *(uint4*)&As[0][arow * 40 + ak]  = av;
        *(uint4*)&Bs[0][brow * 72 + bnc] = bv;
        __syncthreads();
        if (k0 + 32 < K) {
            __builtin_prefetch(Ag + k0 + 32, 0, 1);
            __builtin_prefetch(Bg + (size_t)(k0 + 32) * N, 0, 1);
        }
        gemm_step(As[0], Bs[0], wm, wn, hl, n16, acc0, acc1);
    }
#endif

    #pragma unroll
    for (int s = 0; s < 2; ++s) {
        const v8f a = s ? acc1 : acc0;
        #pragma unroll
        for (int r = 0; r < 8; ++r) {
            const int row = bm + wm * 16 + r + 8 * hl;     // C/D layout
            const int col = bn + wn * 32 + s * 16 + n16;
            float v = a[r] + bias[col];
            if (RELU) v = fmaxf(v, 0.0f);
            const size_t idx = (size_t)row * N + col;
            if (RESID) v += resid[idx];
            if (OUT_BF16) outB[idx] = f32_to_bf16(v);
            else          outF[idx] = v;
        }
    }
}

// ---------------------------------------------------------------------------
// Flash attention (causal). qkv is bf16 [BT, 3C]; head h at cols h*64 (+0/
// +1024/+2048 for Q/K/V). Each wave owns 16 query rows; block = 8 waves =
// 128 rows. Streams 32 keys/step with online softmax; P transposed through
// per-wave LDS to A-layout for the P@V WMMAs.
// ---------------------------------------------------------------------------
__global__ __launch_bounds__(256, 2) void flash_attn_kernel(
    const unsigned short* __restrict__ qkv, float* __restrict__ out) {
    __shared__ unsigned short Plds[8 * 16 * 36];   // per-wave 16x32 bf16, pad

    const int tid = threadIdx.x, lane = tid & 31, w = tid >> 5;
    const int hl = lane >> 4, n16 = lane & 15;
    const int qbase = blockIdx.x * 128 + w * 16;
    const int bh = blockIdx.y, b = bh >> 4, hh = bh & 15;
    const size_t base = (size_t)b * T_LEN * C3 + hh * HDIM;
    unsigned short* P = &Plds[w * 16 * 36];
    const unsigned int* q32 = (const unsigned int*)qkv;

    // Q fragments for K-chunks [0,32) and [32,64): row t = qbase + n16
    BF16x16 q0, q1;
    {
        const size_t qrow = base + (size_t)(qbase + n16) * C3;
        #pragma unroll
        for (int i = 0; i < 8; ++i) {
            const int k0 = koffA(i, hl);
            q0.u32[i] = q32[(qrow + k0) >> 1];
            q1.u32[i] = q32[(qrow + 32 + k0) >> 1];
        }
    }

    v8f o0 = {}, o1 = {}, o2 = {}, o3 = {};
    float mrow[8], lrow[8];
    #pragma unroll
    for (int r = 0; r < 8; ++r) { mrow[r] = -3.0e38f; lrow[r] = 0.0f; }

    const int jmax = (qbase + 15) >> 5;
    for (int j = 0; j <= jmax; ++j) {
        const int kb = j * 32;

        // K^T B-fragments: [kchunk][ntile]; B row kk = hl*16+2i, col = key
        BF16x16 bk00, bk01, bk10, bk11;
        #pragma unroll
        for (int s = 0; s < 2; ++s) {
            const int key = kb + s * 16 + n16;
            const size_t krow = base + EMBED + (size_t)key * C3;
            #pragma unroll
            for (int i = 0; i < 8; ++i) {
                const int kk = hl * 16 + 2 * i;      // contiguous bf16 pair
                const unsigned int lo = q32[(krow + kk) >> 1];
                const unsigned int hi = q32[(krow + 32 + kk) >> 1];
                if (s == 0) { bk00.u32[i] = lo; bk10.u32[i] = hi; }
                else        { bk01.u32[i] = lo; bk11.u32[i] = hi; }
            }
        }

        v8f s0 = {}, s1 = {};
        s0 = wmma_bf16(q0.v, bk00.v, s0);
        s0 = wmma_bf16(q1.v, bk10.v, s0);
        s1 = wmma_bf16(q0.v, bk01.v, s1);
        s1 = wmma_bf16(q1.v, bk11.v, s1);

        // scale + causal mask (score elem: row r+8*hl, key col n16)
        #pragma unroll
        for (int r = 0; r < 8; ++r) {
            const int tq = qbase + r + 8 * hl;
            float v0 = s0[r] * 0.125f;               // 1/sqrt(64)
            float v1 = s1[r] * 0.125f;
            if (kb + n16 > tq)       v0 = -3.0e38f;
            if (kb + 16 + n16 > tq)  v1 = -3.0e38f;
            s0[r] = v0; s1[r] = v1;
        }

        // online softmax per row (reduce over 16-lane half-groups)
        float fsc[8];
        #pragma unroll
        for (int r = 0; r < 8; ++r) {
            float mx = fmaxf(s0[r], s1[r]);
            #pragma unroll
            for (int m = 8; m >= 1; m >>= 1) mx = fmaxf(mx, __shfl_xor(mx, m, 32));
            const float mnew = fmaxf(mrow[r], mx);
            const float f  = __expf(mrow[r] - mnew);
            const float p0 = __expf(s0[r] - mnew);
            const float p1 = __expf(s1[r] - mnew);
            float ps = p0 + p1;
            #pragma unroll
            for (int m = 8; m >= 1; m >>= 1) ps += __shfl_xor(ps, m, 32);
            lrow[r] = lrow[r] * f + ps;
            mrow[r] = mnew;
            fsc[r] = f;
            s0[r] = p0; s1[r] = p1;
        }
        #pragma unroll
        for (int r = 0; r < 8; ++r) {
            o0[r] *= fsc[r]; o1[r] *= fsc[r];
            o2[r] *= fsc[r]; o3[r] *= fsc[r];
        }

        // P (C-layout) -> per-wave LDS -> A-layout fragment
        #pragma unroll
        for (int r = 0; r < 8; ++r) {
            const int row = r + 8 * hl;
            P[row * 36 + n16]      = f32_to_bf16(s0[r]);
            P[row * 36 + 16 + n16] = f32_to_bf16(s1[r]);
        }
        asm volatile("s_wait_dscnt 0" ::: "memory");  // intra-wave LDS RAW
        BF16x16 pf;
        #pragma unroll
        for (int i = 0; i < 8; ++i)
            pf.u32[i] = *(const unsigned int*)&P[n16 * 36 + koffA(i, hl)];

        // P @ V : V B-fragments, 4 hd-subtiles
        #pragma unroll
        for (int s = 0; s < 4; ++s) {
            BF16x16 bv;
            #pragma unroll
            for (int i = 0; i < 8; ++i) {
                const int kk = hl * 16 + 2 * i;      // key index within chunk
                const size_t v0 = base + 2 * EMBED + (size_t)(kb + kk) * C3 + s * 16 + n16;
                const size_t v1 = v0 + C3;
                bv.u16[2 * i]     = qkv[v0];
                bv.u16[2 * i + 1] = qkv[v1];
            }
            if (s == 0) o0 = wmma_bf16(pf.v, bv.v, o0);
            if (s == 1) o1 = wmma_bf16(pf.v, bv.v, o1);
            if (s == 2) o2 = wmma_bf16(pf.v, bv.v, o2);
            if (s == 3) o3 = wmma_bf16(pf.v, bv.v, o3);
        }
    }

    // normalize by l and write fp32 [B,T,C]
    #pragma unroll
    for (int r = 0; r < 8; ++r) {
        const float inv = 1.0f / lrow[r];
        const int tq = qbase + r + 8 * hl;
        const size_t ob = ((size_t)b * T_LEN + tq) * EMBED + hh * HDIM;
        out[ob + n16]      = o0[r] * inv;
        out[ob + 16 + n16] = o1[r] * inv;
        out[ob + 32 + n16] = o2[r] * inv;
        out[ob + 48 + n16] = o3[r] * inv;
    }
}

// ---------------------------------------------------------------------------
extern "C" void kernel_launch(void* const* d_in, const int* in_sizes, int n_in,
                              void* d_out, int out_size, void* d_ws, size_t ws_size,
                              hipStream_t stream) {
    const float* x      = (const float*)d_in[0];
    const float* ln1_s  = (const float*)d_in[1];
    const float* ln1_b  = (const float*)d_in[2];
    const float* w_qkv  = (const float*)d_in[3];
    const float* b_qkv  = (const float*)d_in[4];
    const float* ln2_s  = (const float*)d_in[5];
    const float* ln2_b  = (const float*)d_in[6];
    const float* w_fc   = (const float*)d_in[7];
    const float* b_fc   = (const float*)d_in[8];
    const float* w_proj = (const float*)d_in[9];
    const float* b_proj = (const float*)d_in[10];
    float* outp = (float*)d_out;

    char* ws = (char*)d_ws;
    size_t off = 0;
    unsigned short* h1_bf    = (unsigned short*)(ws + off); off += (size_t)BT * EMBED * 2;
    unsigned short* wqkv_bf  = (unsigned short*)(ws + off); off += (size_t)EMBED * C3 * 2;
    unsigned short* qkv_bf   = (unsigned short*)(ws + off); off += (size_t)BT * C3 * 2;
    float*          attnF    = (float*)(ws + off);          off += (size_t)BT * EMBED * 4;
    float*          x1       = (float*)(ws + off);          off += (size_t)BT * EMBED * 4;
    unsigned short* h2_bf    = (unsigned short*)(ws + off); off += (size_t)BT * EMBED * 2;
    unsigned short* wfc_bf   = (unsigned short*)(ws + off); off += (size_t)EMBED * HMLP * 2;
    unsigned short* wproj_bf = (unsigned short*)(ws + off); off += (size_t)HMLP * EMBED * 2;
    unsigned short* hfc_bf   = (unsigned short*)(ws + off); off += (size_t)BT * HMLP * 2;
    (void)ws_size; (void)in_sizes; (void)n_in; (void)out_size;

    // 1) bf16 weight casts
    cast_bf16_kernel<<<2048, 256, 0, stream>>>(w_qkv,  wqkv_bf,  EMBED * C3);
    cast_bf16_kernel<<<2048, 256, 0, stream>>>(w_fc,   wfc_bf,   EMBED * HMLP);
    cast_bf16_kernel<<<2048, 256, 0, stream>>>(w_proj, wproj_bf, HMLP * EMBED);

    // 2) LN1 -> bf16
    ln_bf16_kernel<<<BT, 256, 0, stream>>>(x, ln1_s, ln1_b, h1_bf);

    // 3) QKV GEMM: [4096,1024]@[1024,3072] + b_qkv -> bf16
    gemm_bf16_kernel<false, true, false><<<dim3(C3 / 64, BT / 64), 256, 0, stream>>>(
        h1_bf, wqkv_bf, b_qkv, nullptr, nullptr, qkv_bf, BT, C3, EMBED);

    // 4) causal flash attention -> fp32
    flash_attn_kernel<<<dim3(T_LEN / 128, 2 * NHEAD), 256, 0, stream>>>(qkv_bf, attnF);

    // 5) x1 = x + attn
    resid_kernel<<<(BT * EMBED) / (256 * 4), 256, 0, stream>>>(x, attnF, x1);

    // 6) LN2 -> bf16
    ln_bf16_kernel<<<BT, 256, 0, stream>>>(x1, ln2_s, ln2_b, h2_bf);

    // 7) MLP fc: [4096,1024]@[1024,4096] + b_fc, ReLU -> bf16
    gemm_bf16_kernel<true, true, false><<<dim3(HMLP / 64, BT / 64), 256, 0, stream>>>(
        h2_bf, wfc_bf, b_fc, nullptr, nullptr, hfc_bf, BT, HMLP, EMBED);

    // 8) MLP proj: [4096,4096]@[4096,1024] + b_proj + x1 -> fp32 out
    gemm_bf16_kernel<false, false, true><<<dim3(EMBED / 64, BT / 64), 256, 0, stream>>>(
        hfc_bf, wproj_bf, b_proj, x1, outp, nullptr, BT, EMBED, HMLP);
}
